// ACOPFEmbedder_Bus_39694087749649
// MI455X (gfx1250) — matrix operand order repaired and checked by reading, not compiled, not run on hardware
//
#include <hip/hip_runtime.h>

// ---------------------------------------------------------------------------
// ACOPF bus embedder for MI455X (gfx1250, wave32).
//
// Cost model: Wfc is 4 x 4096 x 16384 f32 = 1.07 GB, read exactly once per
// call -> bandwidth floor ~46us @ 23.3 TB/s. Everything else is <= a few MB.
// So: attention phase must avoid atomics (CSR + per-node online softmax),
// and the FC GEMV must stream Wfc at full width; we run it through
// v_wmma_f32_16x16x32_f16 (A = broadcast val chunk, B = Wfc^T tile) so the
// dominant loop uses the CDNA5 matrix path with f32 accumulation.
// ---------------------------------------------------------------------------

#define T_ 4
#define L_ 4
#define N_ 2048
#define E_ 65536
#define H_ 64
#define C_ 2
#define ROWS_ (2 * N_)            // 4096 FC output rows per type
#define KDIM_ (2 * L_ * N_)       // 16384 FC inner dim per type
#define KSPLIT_ 4
#define KS_ (KDIM_ / KSPLIT_)     // 4096 K per split

typedef _Float16 v16h __attribute__((ext_vector_type(16)));
typedef float    v8f  __attribute__((ext_vector_type(8)));

// ---------------- CSR build (graph is shared across the L layers) ----------

__global__ void k_init_counts(int* __restrict__ counts) {
  int i = blockIdx.x * blockDim.x + threadIdx.x;
  if (i < T_ * N_) counts[i] = 0;
}

__global__ void k_count(const int* __restrict__ tgt, int* __restrict__ counts) {
  int i = blockIdx.x * blockDim.x + threadIdx.x;
  if (i >= T_ * E_) return;
  int t = i / E_;
  atomicAdd(&counts[t * N_ + tgt[i]], 1);
}

// one block (256 threads) per type: exclusive scan of 2048 counts via LDS
__global__ void k_scan(const int* __restrict__ counts,
                       int* __restrict__ offsets, int* __restrict__ cursor) {
  __shared__ int lsum[256];
  const int PER = N_ / 256;  // 8
  int t = blockIdx.x, tid = threadIdx.x;
  int base = t * N_ + tid * PER;
  int loc[PER];
  int run = 0;
  #pragma unroll
  for (int j = 0; j < PER; ++j) { loc[j] = run; run += counts[base + j]; }
  lsum[tid] = run;
  __syncthreads();
  for (int off = 1; off < 256; off <<= 1) {
    int add = (tid >= off) ? lsum[tid - off] : 0;
    __syncthreads();
    lsum[tid] += add;
    __syncthreads();
  }
  int cbase = (tid > 0) ? lsum[tid - 1] : 0;
  #pragma unroll
  for (int j = 0; j < PER; ++j) {
    int o = cbase + loc[j];
    offsets[t * (N_ + 1) + tid * PER + j] = o;
    cursor[base + j] = o;
  }
  if (tid == 255) offsets[t * (N_ + 1) + N_] = lsum[255];
}

__global__ void k_fill(const int* __restrict__ tgt, int* __restrict__ cursor,
                       int* __restrict__ perm) {
  int i = blockIdx.x * blockDim.x + threadIdx.x;
  if (i >= T_ * E_) return;
  int t = i / E_, e = i % E_;
  int pos = atomicAdd(&cursor[t * N_ + tgt[i]], 1);
  perm[t * E_ + pos] = e;
}

// ---------------- fused TransformerConv + head-sum -------------------------
// one wave32 per (t, l, n); lane owns heads h=lane and h=lane+32.
// Online softmax over the node's edge list (exact same math as two-pass
// max/exp/sum of the reference, fp-reassociated). Zero atomics.

__global__ void k_attn(const float* __restrict__ x,
                       const float* __restrict__ ea,
                       const int* __restrict__ src,
                       const float* __restrict__ Wq, const float* __restrict__ bq,
                       const float* __restrict__ Wk, const float* __restrict__ bk,
                       const float* __restrict__ Wv, const float* __restrict__ bv,
                       const float* __restrict__ We,
                       const float* __restrict__ Wsk, const float* __restrict__ bsk,
                       const int* __restrict__ offsets, const int* __restrict__ counts,
                       const int* __restrict__ perm,
                       float* __restrict__ val) {
  int gw = (blockIdx.x * blockDim.x + threadIdx.x) >> 5;
  int lane = threadIdx.x & 31;
  int t = gw / (L_ * N_);
  int r = gw % (L_ * N_);
  int l = r / N_;
  int n = r % N_;

  int wb = (t * L_ + l) * (C_ * H_);  // [C=2][H=64] weight block
  int bb = (t * L_ + l) * H_;

  float x0 = x[(t * N_ + n) * C_ + 0];
  float x1 = x[(t * N_ + n) * C_ + 1];

  float q[2], wk0[2], wk1[2], bkv[2], wv0[2], wv1[2], bvv[2], we0[2], we1[2], sk[2];
  #pragma unroll
  for (int j = 0; j < 2; ++j) {
    int h = lane + 32 * j;
    q[j]  = x0 * Wq[wb + h] + x1 * Wq[wb + H_ + h] + bq[bb + h];
    wk0[j] = Wk[wb + h];  wk1[j] = Wk[wb + H_ + h];  bkv[j] = bk[bb + h];
    wv0[j] = Wv[wb + h];  wv1[j] = Wv[wb + H_ + h];  bvv[j] = bv[bb + h];
    we0[j] = We[wb + h];  we1[j] = We[wb + H_ + h];
    sk[j] = x0 * Wsk[wb + h] + x1 * Wsk[wb + H_ + h] + bsk[bb + h];
  }

  int start = offsets[t * (N_ + 1) + n];
  int cnt   = counts[t * N_ + n];
  const int*   pe  = perm + t * E_ + start;
  const int*   ps  = src + t * E_;
  const float* eat = ea + (size_t)t * E_ * C_;

  float M = -__builtin_inff(), S = 0.f, m0 = 0.f, m1 = 0.f;
  for (int i = 0; i < cnt; ++i) {
    int e = pe[i];
    int s = ps[e];                       // index into x_all = x.reshape(T*N, C)
    float xs0 = x[s * 2], xs1 = x[s * 2 + 1];
    float e0 = eat[e * 2], e1 = eat[e * 2 + 1];
    float k0 = xs0 * wk0[0] + xs1 * wk1[0] + bkv[0] + e0 * we0[0] + e1 * we1[0];
    float k1 = xs0 * wk0[1] + xs1 * wk1[1] + bkv[1] + e0 * we0[1] + e1 * we1[1];
    float part = q[0] * k0 + q[1] * k1;
    #pragma unroll
    for (int o = 16; o > 0; o >>= 1) part += __shfl_xor(part, o, 32);
    float alpha = part * 0.125f;         // 1/sqrt(H=64)
    float nM = fmaxf(M, alpha);
    float sc = __expf(M - nM);           // first edge: exp(-inf)=0
    float w  = __expf(alpha - nM);
    float v0 = xs0 * wv0[0] + xs1 * wv1[0] + bvv[0] + e0 * we0[0] + e1 * we1[0];
    float v1 = xs0 * wv0[1] + xs1 * wv1[1] + bvv[1] + e0 * we0[1] + e1 * we1[1];
    S  = S  * sc + w;
    m0 = m0 * sc + w * v0;
    m1 = m1 * sc + w * v1;
    M = nM;
  }
  float inv = 1.f / (S + 1e-16f);        // cnt==0 -> msg=0, out=skip (matches ref)
  float o0 = m0 * inv + sk[0];           // head h=lane      (< 32) -> P
  float o1 = m1 * inv + sk[1];           // head h=lane+32   (>=32) -> Q
  #pragma unroll
  for (int o = 16; o > 0; o >>= 1) {
    o0 += __shfl_xor(o0, o, 32);
    o1 += __shfl_xor(o1, o, 32);
  }
  if (lane == 0) {
    size_t rr = (size_t)t * KDIM_ + (size_t)(l * N_ + n) * 2;
    val[rr + 0] = o0;                    // P
    val[rr + 1] = o1;                    // Q
  }
}

// ---------------- FC GEMV through WMMA f16 16x16x32 ------------------------
// A[m,k] = val[kb+k] broadcast over all 16 M rows -> D[m,n] replicated over m,
// so lane l<16 holds y_partial[rowbase+l] in acc[0] per the documented D
// layout. B[k,n] = Wfc[rowbase+n, kb+k]: lane = n (+16 selects K half), each
// lane loads 16 consecutive f32 (4x B128); lanes l & l+16 consume one 128B
// line of the row. f32 accumulation, K split 4-way into partial buffers.

__global__ void k_fc(const float* __restrict__ Wfc,
                     const float* __restrict__ val,
                     float* __restrict__ partial) {
  int gw = (blockIdx.x * blockDim.x + threadIdx.x) >> 5;
  int lane = threadIdx.x & 31;
  const int TILES = ROWS_ / 16;          // 256
  int t  = gw / (TILES * KSPLIT_);
  int r  = gw % (TILES * KSPLIT_);
  int rt = r / KSPLIT_;
  int ks = r % KSPLIT_;
  int rowbase = rt * 16;
  int kb0 = ks * KS_;
  int hsel = lane >> 4;                  // K-half selector
  int nn   = lane & 15;                  // output row within tile (B: N index)

  const float* vt = val + (size_t)t * KDIM_;
  const float* wrow = Wfc + ((size_t)(t * ROWS_ + rowbase + nn)) * KDIM_;

  v8f acc = {0.f, 0.f, 0.f, 0.f, 0.f, 0.f, 0.f, 0.f};

  for (int i = 0; i < KS_ / 32; ++i) {
    int kb = kb0 + i * 32;
    // A chunks: K in [16h, 16h+8) and [16h+8, 16h+16)
    const float4* ap0 = (const float4*)(vt + kb + 16 * hsel);
    const float4* ap1 = (const float4*)(vt + kb + 16 * hsel + 8);
    float4 a0 = ap0[0], a1 = ap0[1], a2 = ap1[0], a3 = ap1[1];
    // B: 16 consecutive K at [16h, 16h+16)
    const float4* bp = (const float4*)(wrow + kb + 16 * hsel);
    __builtin_prefetch((const void*)(wrow + kb + 16 * hsel + 256), 0, 1);
    float4 b0 = bp[0], b1 = bp[1], b2 = bp[2], b3 = bp[3];

    v16h A, B;
    A[0] = (_Float16)a0.x;  A[1] = (_Float16)a0.y;
    A[2] = (_Float16)a0.z;  A[3] = (_Float16)a0.w;
    A[4] = (_Float16)a1.x;  A[5] = (_Float16)a1.y;
    A[6] = (_Float16)a1.z;  A[7] = (_Float16)a1.w;
    A[8] = (_Float16)a2.x;  A[9] = (_Float16)a2.y;
    A[10] = (_Float16)a2.z; A[11] = (_Float16)a2.w;
    A[12] = (_Float16)a3.x; A[13] = (_Float16)a3.y;
    A[14] = (_Float16)a3.z; A[15] = (_Float16)a3.w;

    B[0] = (_Float16)b0.x;  B[1] = (_Float16)b0.y;
    B[2] = (_Float16)b0.z;  B[3] = (_Float16)b0.w;
    B[4] = (_Float16)b1.x;  B[5] = (_Float16)b1.y;
    B[6] = (_Float16)b1.z;  B[7] = (_Float16)b1.w;
    B[8] = (_Float16)b2.x;  B[9] = (_Float16)b2.y;
    B[10] = (_Float16)b2.z; B[11] = (_Float16)b2.w;
    B[12] = (_Float16)b3.x; B[13] = (_Float16)b3.y;
    B[14] = (_Float16)b3.z; B[15] = (_Float16)b3.w;

    acc = __builtin_amdgcn_wmma_f32_16x16x32_f16(
        false, A, false, B, (short)0, acc, false, false);
  }

  if (lane < 16) {
    partial[(size_t)ks * (T_ * ROWS_) + t * ROWS_ + rowbase + lane] = acc[0];
  }
}

__global__ void k_reduce(const float* __restrict__ partial,
                         const float* __restrict__ bfc,
                         float* __restrict__ out) {
  int i = blockIdx.x * blockDim.x + threadIdx.x;
  if (i >= T_ * ROWS_) return;
  float s = bfc[i];
  #pragma unroll
  for (int ks = 0; ks < KSPLIT_; ++ks) s += partial[(size_t)ks * (T_ * ROWS_) + i];
  out[i] = s;  // out[t, n, c] flat == y[t, j] flat, j = n*2 + c
}

// ---------------------------------------------------------------------------

extern "C" void kernel_launch(void* const* d_in, const int* in_sizes, int n_in,
                              void* d_out, int out_size, void* d_ws, size_t ws_size,
                              hipStream_t stream) {
  (void)in_sizes; (void)n_in; (void)out_size; (void)ws_size;
  const float* x    = (const float*)d_in[0];
  const float* ea   = (const float*)d_in[1];
  const int*   src  = (const int*)d_in[2];
  const int*   tgt  = (const int*)d_in[3];
  const float* Wq   = (const float*)d_in[4];
  const float* bq   = (const float*)d_in[5];
  const float* Wk   = (const float*)d_in[6];
  const float* bk   = (const float*)d_in[7];
  const float* Wv   = (const float*)d_in[8];
  const float* bv   = (const float*)d_in[9];
  const float* We   = (const float*)d_in[10];
  const float* Wsk  = (const float*)d_in[11];
  const float* bsk  = (const float*)d_in[12];
  const float* Wfc  = (const float*)d_in[13];
  const float* bfc  = (const float*)d_in[14];
  float* out = (float*)d_out;

  // workspace layout (bytes, 16B-aligned chunks)
  char* ws = (char*)d_ws;
  int*   counts  = (int*)(ws);                          // 8192  ints
  int*   offsets = (int*)(ws + 32768);                  // 8196  ints
  int*   cursor  = (int*)(ws + 65552);                  // 8192  ints
  int*   perm    = (int*)(ws + 98320);                  // 262144 ints
  float* val     = (float*)(ws + 1146896);              // 65536 f32
  float* partial = (float*)(ws + 1409040);              // 65536 f32
  // total ~1.6 MB

  k_init_counts<<<(T_ * N_ + 255) / 256, 256, 0, stream>>>(counts);
  k_count<<<(T_ * E_) / 256, 256, 0, stream>>>(tgt, counts);
  k_scan<<<T_, 256, 0, stream>>>(counts, offsets, cursor);
  k_fill<<<(T_ * E_) / 256, 256, 0, stream>>>(tgt, cursor, perm);

  // 1 wave per (t,l,n): 32768 waves, 8 waves/block
  k_attn<<<(T_ * L_ * N_) / 8, 256, 0, stream>>>(
      x, ea, src, Wq, bq, Wk, bk, Wv, bv, We, Wsk, bsk,
      offsets, counts, perm, val);

  // 4 types * 256 row-tiles * 4 K-splits = 4096 waves, 8 waves/block
  k_fc<<<(T_ * (ROWS_ / 16) * KSPLIT_) / 8, 256, 0, stream>>>(Wfc, val, partial);

  k_reduce<<<(T_ * ROWS_ + 255) / 256, 256, 0, stream>>>(partial, bfc, out);
}
